// SelfAttentionLayer2_21380347200300
// MI455X (gfx1250) — compile-verified, matrix operand order checked by reading, not run on hardware
//
#include <hip/hip_runtime.h>
#include <hip/hip_bf16.h>
#include <cstdint>

typedef _Float16 half_t;
typedef __attribute__((ext_vector_type(16))) _Float16 v16h;
typedef __attribute__((ext_vector_type(8)))  _Float16 v8h;
typedef __attribute__((ext_vector_type(8)))  float    v8f;

// ---------------------------------------------------------------------------
// WMMA helper (CDNA5 wave32, v_wmma_f32_16x16x32_f16)
// ---------------------------------------------------------------------------
__device__ __forceinline__ v8f wmma_f16(v16h a, v16h b, v8f c) {
  // (neg_a, A, neg_b, B, c_mod, C, reuse_a, reuse_b)
  return __builtin_amdgcn_wmma_f32_16x16x32_f16(false, a, false, b, (short)0, c,
                                                false, false);
}

// Load a 16x32 fragment (A-style) from a row-major f16 matrix with leading
// dimension `ld` (elements). Same gather serves B fragments of A*W^T (W
// row-major [out][in]), Q*K^T (K row-major) and P*V (V^T row-major).
// CDNA5 layout: row = lane&15; element e -> k = (e&7) + 16*(e>>3) + 8*(lane>>4).
__device__ __forceinline__ v16h frag_ld(const half_t* base, int ld, int lane) {
  const int row = lane & 15;
  const int hf  = (lane >> 4) & 1;
  const half_t* p = base + (size_t)row * ld + 8 * hf;
  v8h lo = *(const v8h*)(p);        // k = 8*hf + 0..7
  v8h hi = *(const v8h*)(p + 16);   // k = 16 + 8*hf + 0..7
  v16h r;
#pragma unroll
  for (int i = 0; i < 8; ++i) { r[i] = lo[i]; r[8 + i] = hi[i]; }
  return r;
}

// ---------------------------------------------------------------------------
// Kernel 0: f32 -> f16 weight cast
// ---------------------------------------------------------------------------
__global__ void cvt_f16_kernel(const float* __restrict__ src,
                               half_t* __restrict__ dst, int n) {
  int i = blockIdx.x * 256 + threadIdx.x;
  if (i < n) dst[i] = (half_t)src[i];
}

// ---------------------------------------------------------------------------
// Kernel 1: LayerNorm + window partition + f16 cast.
// One wave per token (C=512 -> 16 elems/lane), 8 waves / block.
// ---------------------------------------------------------------------------
__global__ void ln_part_kernel(const float* __restrict__ tgt,
                               const float* __restrict__ gamma,
                               const float* __restrict__ beta,
                               half_t* __restrict__ xh,
                               const int* __restrict__ indp) {
  const int ind  = *indp;
  const int tok  = blockIdx.x * 8 + (threadIdx.x >> 5);
  const int lane = threadIdx.x & 31;
  const float* x = tgt + (size_t)tok * 512;

  float v[16];
  float s = 0.f;
#pragma unroll
  for (int j = 0; j < 16; ++j) { v[j] = x[j * 32 + lane]; s += v[j]; }
#pragma unroll
  for (int m = 16; m >= 1; m >>= 1) s += __shfl_xor(s, m, 32);
  const float mean = s * (1.f / 512.f);

  float q = 0.f;
#pragma unroll
  for (int j = 0; j < 16; ++j) { float d = v[j] - mean; q += d * d; }
#pragma unroll
  for (int m = 16; m >= 1; m >>= 1) q += __shfl_xor(q, m, 32);
  const float inv = rsqrtf(q * (1.f / 512.f) + 1e-5f);

  // partition: token = (b, h,p1,w,p2,d,p3) base-4 digits
  const int bb = tok >> 12, r = tok & 4095;
  const int h  = (r >> 10) & 3, p1 = (r >> 8) & 3, w = (r >> 6) & 3;
  const int p2 = (r >> 4) & 3,  d  = (r >> 2) & 3, p3 = r & 3;
  int winpos;
  if (ind & 1)
    winpos = (bb * 64 + (p1 * 16 + p2 * 4 + p3)) * 64 + (h * 16 + w * 4 + d);
  else
    winpos = (bb * 64 + (h * 16 + w * 4 + d)) * 64 + (p1 * 16 + p2 * 4 + p3);

  half_t* dst = xh + (size_t)winpos * 512;
#pragma unroll
  for (int j = 0; j < 16; ++j) {
    const int c = j * 32 + lane;
    dst[c] = (half_t)((v[j] - mean) * inv * gamma[c] + beta[c]);
  }
}

// ---------------------------------------------------------------------------
// Kernel 2: per-(window, head) fused QKV + attention.
// Block = 128 threads (4 waves); wave w owns row-strip tm = w.
// Grid = 1024 windows * 8 heads.
// ---------------------------------------------------------------------------
__global__ void attn_kernel(const half_t* __restrict__ xh,
                            const half_t* __restrict__ wqkv,
                            half_t* __restrict__ oh, float scale) {
  const int win  = blockIdx.x >> 3;
  const int head = blockIdx.x & 7;
  const int tid  = threadIdx.x;
  const int wave = tid >> 5;          // == tm row strip
  const int lane = tid & 31;

  __shared__ half_t Qs[64 * 64];      // Q row-major [n][d] (later reused as Os)
  __shared__ half_t Ks[64 * 64];      // K row-major [n][d]
  __shared__ half_t Vts[64 * 64];     // V transposed [d][m]
  __shared__ float  Ss[64 * 64];      // scores f32
  __shared__ half_t Ps[64 * 64];      // softmax probs f16

  const half_t* Xw = xh + (size_t)win * 64 * 512;
  const half_t* Wh = wqkv + (size_t)head * 64 * 512;   // offset within each matrix
  const int col16  = lane & 15;
  const int rbase0 = 8 * ((lane >> 4) & 1);
  const int tm     = wave;

  // ---- QKV: one A-fragment feeds 12 tiles (3 matrices x 4 tn), K=512 ----
  v8f acc[12];
#pragma unroll
  for (int j = 0; j < 12; ++j) acc[j] = (v8f){};
#pragma unroll
  for (int kk = 0; kk < 16; ++kk) {
    const half_t* ap = Xw + tm * 16 * 512 + kk * 32;
    if (kk < 12) __builtin_prefetch(ap + 128, 0, 1);   // global_prefetch_b8
    v16h a = frag_ld(ap, 512, lane);
#pragma unroll
    for (int j = 0; j < 12; ++j) {
      const int mat = j >> 2;        // 0=Q 1=K 2=V
      const int tn  = j & 3;
      const half_t* Wb = Wh + (size_t)(mat * 512 + tn * 16) * 512 + kk * 32;
      v16h b = frag_ld(Wb, 512, lane);
      acc[j] = wmma_f16(a, b, acc[j]);
    }
  }
#pragma unroll
  for (int j = 0; j < 12; ++j) {
    const int mat = j >> 2, tn = j & 3;
    const int col = tn * 16 + col16;
    const int rb  = tm * 16 + rbase0;
    if (mat == 0) {
#pragma unroll
      for (int i = 0; i < 8; ++i) Qs[(rb + i) * 64 + col] = (half_t)acc[j][i];
    } else if (mat == 1) {
#pragma unroll
      for (int i = 0; i < 8; ++i) Ks[(rb + i) * 64 + col] = (half_t)acc[j][i];
    } else {
      v8h pk;
#pragma unroll
      for (int i = 0; i < 8; ++i) pk[i] = (half_t)acc[j][i];
      *(v8h*)(Vts + col * 64 + rb) = pk;   // transposed, contiguous 16B
    }
  }
  __syncthreads();

  // ---- S = (Q K^T) * scale : wave owns row strip tm; A pair reused 4x ----
  {
    v16h a0 = frag_ld(Qs + tm * 16 * 64, 64, lane);
    v16h a1 = frag_ld(Qs + tm * 16 * 64 + 32, 64, lane);
#pragma unroll
    for (int tn = 0; tn < 4; ++tn) {
      v16h b0 = frag_ld(Ks + tn * 16 * 64, 64, lane);
      v16h b1 = frag_ld(Ks + tn * 16 * 64 + 32, 64, lane);
      v8f s = {};
      s = wmma_f16(a0, b0, s);
      s = wmma_f16(a1, b1, s);
      const int col = tn * 16 + col16;
      const int rb  = tm * 16 + rbase0;
#pragma unroll
      for (int i = 0; i < 8; ++i) Ss[(rb + i) * 64 + col] = s[i] * scale;
    }
  }
  __syncthreads();

  // ---- row softmax: 2 lanes per row, combine with shfl_xor ----
  {
    const int row  = tid >> 1;
    const int part = tid & 1;
    const float* r = Ss + row * 64 + part * 32;
    float mx = -3.0e38f;
#pragma unroll
    for (int j = 0; j < 32; ++j) mx = fmaxf(mx, r[j]);
    mx = fmaxf(mx, __shfl_xor(mx, 1, 32));
    float sum = 0.f;
#pragma unroll
    for (int j = 0; j < 32; ++j) sum += __expf(r[j] - mx);
    sum += __shfl_xor(sum, 1, 32);
    const float inv = 1.f / sum;
    half_t* pr = Ps + row * 64 + part * 32;
#pragma unroll
    for (int j = 0; j < 32; ++j) pr[j] = (half_t)(__expf(r[j] - mx) * inv);
  }
  __syncthreads();

  // ---- O = P V : A pair reused 4x; stage result in LDS (reuse Qs) ----
  half_t* Os = Qs;  // Q no longer needed
  {
    v16h a0 = frag_ld(Ps + tm * 16 * 64, 64, lane);
    v16h a1 = frag_ld(Ps + tm * 16 * 64 + 32, 64, lane);
#pragma unroll
    for (int tn = 0; tn < 4; ++tn) {
      v16h b0 = frag_ld(Vts + tn * 16 * 64, 64, lane);
      v16h b1 = frag_ld(Vts + tn * 16 * 64 + 32, 64, lane);
      v8f o = {};
      o = wmma_f16(a0, b0, o);
      o = wmma_f16(a1, b1, o);
      const int col = tn * 16 + col16;
      const int rb  = tm * 16 + rbase0;
#pragma unroll
      for (int i = 0; i < 8; ++i) Os[(rb + i) * 64 + col] = (half_t)o[i];
    }
  }
  __syncthreads();

  // ---- coalesced 16B row stores of the head output ----
  half_t* ohw = oh + (size_t)win * 64 * 512 + head * 64;
  for (int idx = tid; idx < 64 * 8; idx += 128) {
    const int row = idx >> 3, c8 = idx & 7;
    *(v8h*)(ohw + (size_t)row * 512 + c8 * 8) = *(const v8h*)(Os + row * 64 + c8 * 8);
  }
}

// ---------------------------------------------------------------------------
// Kernel 3: out-proj + bias + un-partition + residual (f32 output).
// Block = 256 threads (8 waves), one block per window.
// wave -> tm = wave&3, tn half = wave>>2; two passes of 8 tn tiles;
// one A-fragment feeds 8 WMMAs per k-step.
// ---------------------------------------------------------------------------
__device__ __forceinline__ int unpart_index(int win, int n, int ind) {
  const int b = win >> 6, wr = win & 63;
  int h, w, d, p1, p2, p3;
  if (ind & 1) {  // window = (p1,p2,p3), pos = (h,w,d)
    p1 = (wr >> 4) & 3; p2 = (wr >> 2) & 3; p3 = wr & 3;
    h  = (n  >> 4) & 3; w  = (n  >> 2) & 3; d  = n & 3;
  } else {        // window = (h,w,d), pos = (p1,p2,p3)
    h  = (wr >> 4) & 3; w  = (wr >> 2) & 3; d  = wr & 3;
    p1 = (n  >> 4) & 3; p2 = (n  >> 2) & 3; p3 = n & 3;
  }
  const int r = ((((h * 4 + p1) * 4 + w) * 4 + p2) * 4 + d) * 4 + p3;
  return b * 4096 + r;
}

__global__ void proj_kernel(const half_t* __restrict__ oh,
                            const half_t* __restrict__ wproj,
                            const float* __restrict__ bproj,
                            const float* __restrict__ tgt,
                            float* __restrict__ out,
                            const int* __restrict__ indp) {
  const int ind  = *indp;
  const int win  = blockIdx.x;
  const int wave = threadIdx.x >> 5;
  const int lane = threadIdx.x & 31;
  const half_t* Ow = oh + (size_t)win * 64 * 512;
  const int col16  = lane & 15;
  const int rbase0 = 8 * ((lane >> 4) & 1);
  const int tm     = wave & 3;
  const int nhalf  = wave >> 2;       // 0 or 1

  for (int pass = 0; pass < 2; ++pass) {
    const int tnb = nhalf * 16 + pass * 8;   // 8 tn tiles per pass
    v8f acc[8];
#pragma unroll
    for (int j = 0; j < 8; ++j) acc[j] = (v8f){};
#pragma unroll
    for (int kk = 0; kk < 16; ++kk) {
      const half_t* ap = Ow + tm * 16 * 512 + kk * 32;
      if (kk < 12) __builtin_prefetch(ap + 128, 0, 1);   // global_prefetch_b8
      v16h a = frag_ld(ap, 512, lane);
#pragma unroll
      for (int j = 0; j < 8; ++j) {
        v16h b = frag_ld(wproj + (size_t)((tnb + j) * 16) * 512 + kk * 32, 512, lane);
        acc[j] = wmma_f16(a, b, acc[j]);
      }
    }
#pragma unroll
    for (int j = 0; j < 8; ++j) {
      const int col  = (tnb + j) * 16 + col16;
      const int rb   = tm * 16 + rbase0;
      const float bias = bproj[col];
#pragma unroll
      for (int i = 0; i < 8; ++i) {
        const int g = unpart_index(win, rb + i, ind);
        const size_t idx = (size_t)g * 512 + col;
        out[idx] = tgt[idx] + acc[j][i] + bias;
      }
    }
  }
}

// ---------------------------------------------------------------------------
// Host launcher
// ---------------------------------------------------------------------------
extern "C" void kernel_launch(void* const* d_in, const int* in_sizes, int n_in,
                              void* d_out, int out_size, void* d_ws, size_t ws_size,
                              hipStream_t stream) {
  const float* tgt    = (const float*)d_in[0];   // [16,4096,512]
  const float* w_qkv  = (const float*)d_in[1];   // [1536,512]
  const float* w_proj = (const float*)d_in[2];   // [512,512]
  const float* b_proj = (const float*)d_in[3];   // [512]
  const float* gamma  = (const float*)d_in[4];   // [512]
  const float* beta   = (const float*)d_in[5];   // [512]
  const int*   indp   = (const int*)d_in[6];     // scalar
  float* out = (float*)d_out;

  const size_t NTOK = 65536;                     // 16*4096
  uint8_t* ws = (uint8_t*)d_ws;
  half_t* xh     = (half_t*)(ws);                                // 64 MB
  half_t* oh     = (half_t*)(ws + NTOK * 512 * sizeof(half_t));  // 64 MB
  half_t* wqkvh  = (half_t*)(ws + 2 * NTOK * 512 * sizeof(half_t));
  half_t* wprojh = (half_t*)((uint8_t*)wqkvh + (size_t)1536 * 512 * sizeof(half_t));

  // weights -> f16
  cvt_f16_kernel<<<(1536 * 512 + 255) / 256, 256, 0, stream>>>(w_qkv, wqkvh,
                                                               1536 * 512);
  cvt_f16_kernel<<<(512 * 512 + 255) / 256, 256, 0, stream>>>(w_proj, wprojh,
                                                              512 * 512);
  // LayerNorm + partition
  ln_part_kernel<<<NTOK / 8, 256, 0, stream>>>(tgt, gamma, beta, xh, indp);

  // per-(window, head) attention
  const float scale = 0.044194173824159216f;  // 512^-0.5 (d_model^-0.5)
  attn_kernel<<<1024 * 8, 128, 0, stream>>>(xh, wqkvh, oh, scale);

  // projection + unpartition + residual
  proj_kernel<<<1024, 256, 0, stream>>>(oh, wprojh, b_proj, tgt, out, indp);
}
